// RNN_43250320670827
// MI455X (gfx1250) — matrix operand order repaired and checked by reading, not compiled, not run on hardware
//
#include <hip/hip_runtime.h>
#include <hip/hip_bf16.h>
#include <math.h>

// ---------------------------------------------------------------------------
// Problem constants (from reference): V=50000, E=300, H=512, O=3, B=512, T=512
// Recurrent GEMM per step: M=B=512, K=E+H=812 (padded to 832=26*32), N=H=512.
// ---------------------------------------------------------------------------
static constexpr int Bsz = 512;
static constexpr int Tsz = 512;
static constexpr int Esz = 300;
static constexpr int Hsz = 512;
static constexpr int Osz = 3;
static constexpr int Ksz = Esz + Hsz;   // 812
static constexpr int KP  = 832;         // padded K (26 * 32)
static constexpr int KC  = KP / 32;     // 26 k-chunks of 32

typedef __bf16 bf16_t;
typedef __attribute__((ext_vector_type(16))) __bf16 v16bf;
typedef __attribute__((ext_vector_type(8)))  __bf16 v8bf;
typedef __attribute__((ext_vector_type(8)))  float  v8f;

__device__ __forceinline__ bf16_t f2bf(float f) {
    union { float f; unsigned u; } in; in.f = f;
    unsigned u = in.u + (0x7FFFu + ((in.u >> 16) & 1u));   // round-to-nearest-even
    union { unsigned short s; bf16_t b; } out; out.s = (unsigned short)(u >> 16);
    return out.b;
}
__device__ __forceinline__ float bf2f(bf16_t b) {
    union { bf16_t b; unsigned short s; } in; in.b = b;
    union { unsigned u; float f; } out; out.u = ((unsigned)in.s) << 16;
    return out.f;
}

__device__ __forceinline__ v16bf load_frag(const bf16_t* p0, const bf16_t* p1) {
    v8bf lo = *reinterpret_cast<const v8bf*>(p0);
    v8bf hi = *reinterpret_cast<const v8bf*>(p1);
    return __builtin_shufflevector(lo, hi, 0,1,2,3,4,5,6,7,8,9,10,11,12,13,14,15);
}

// ---------------------------------------------------------------------------
// Init: build Wb = W_i2h as bf16 [H][KP] (zero-padded cols 812..831),
// build A0 = [emb_0 | h0=0 | pad=0] bf16 [B][KP], zero pad columns of A1.
// ---------------------------------------------------------------------------
__global__ void rnn_init_kernel(const int* __restrict__ batch_x,
                                const float* __restrict__ emb_table,
                                const float* __restrict__ W_i2h,
                                bf16_t* __restrict__ Wb,
                                bf16_t* __restrict__ A0,
                                bf16_t* __restrict__ A1) {
    const int tid  = blockIdx.x * blockDim.x + threadIdx.x;
    const int nthr = gridDim.x * blockDim.x;

    // Wb[n][k] = (k < K) ? W_i2h[n][k] : 0
    for (int i = tid; i < KP * Hsz; i += nthr) {
        int k = i / Hsz, n = i - k * Hsz;
        float v = (k < Ksz) ? W_i2h[(size_t)n * Ksz + k] : 0.0f;
        Wb[(size_t)n * KP + k] = f2bf(v);
    }
    // A0 = [emb(t=0) | zeros]
    for (int i = tid; i < Bsz * KP; i += nthr) {
        int b = i / KP, c = i - b * KP;
        float v = 0.0f;
        if (c < Esz) {
            int tok = batch_x[(size_t)b * Tsz + 0];
            v = emb_table[(size_t)tok * Esz + c];
        }
        A0[(size_t)b * KP + c] = f2bf(v);
    }
    // A1 pad columns [K, KP) = 0 (never touched by step kernels)
    for (int i = tid; i < Bsz * (KP - Ksz); i += nthr) {
        int b = i / (KP - Ksz), c = Ksz + (i - b * (KP - Ksz));
        A1[(size_t)b * KP + c] = f2bf(0.0f);
    }
}

// ---------------------------------------------------------------------------
// One recurrence step: Anext[:,E:E+H] = bf16(Acur @ Wb + b_i2h)
//                      Anext[:,0:E]   = bf16(emb_table[batch_x[:, t_next]])
// Grid 4x8 blocks, 256 threads (8 waves). Block tile 64(M)x128(N).
// Wave w: wm=w&1 (2 M groups of 32 rows), wn=w>>1 (4 N groups of 32 cols).
// Each wave: 32x32 tile = 2x2 WMMA tiles, 4 f32 accumulators, 26 k32 chunks.
// Per chunk: 2 A frags + 2 B frags (8x b128 loads) -> 4 WMMAs (2 loads/WMMA).
// ---------------------------------------------------------------------------
__global__ __launch_bounds__(256)
void rnn_step_kernel(const bf16_t* __restrict__ Acur,
                     bf16_t* __restrict__ Anext,
                     const bf16_t* __restrict__ Wb,
                     const float* __restrict__ bias,
                     const int* __restrict__ batch_x,
                     const float* __restrict__ emb_table,
                     int t_next) {
    const int tid  = threadIdx.x;
    const int lane = tid & 31;
    const int wave = tid >> 5;

    // ---- fused gather of next timestep's embeddings into Anext[:,0:E) ----
    {
        const int gtid = (blockIdx.y * gridDim.x + blockIdx.x) * blockDim.x + tid;
        const int gn   = gridDim.x * gridDim.y * blockDim.x;
        for (int i = gtid; i < Bsz * Esz; i += gn) {
            int b = i / Esz, e = i - b * Esz;
            int tok = batch_x[(size_t)b * Tsz + t_next];
            Anext[(size_t)b * KP + e] = f2bf(emb_table[(size_t)tok * Esz + e]);
        }
    }

    // ---- WMMA GEMM: 32x32 output tile per wave ----
    const int wm    = wave & 1;
    const int wn    = wave >> 1;
    const int mbase = blockIdx.y * 64 + wm * 32;     // first of 32 output rows
    const int nb    = blockIdx.x * 128 + wn * 32;    // first of 32 output cols
    const int arow0 = mbase + (lane & 15);           // A row for m-tile 0
    const int asub  = (lane >> 4) * 8;               // K sub-offset (A layout)

    v8f acc00 = {};  // (m0, n0)
    v8f acc01 = {};  // (m0, n1)
    v8f acc10 = {};  // (m1, n0)
    v8f acc11 = {};  // (m1, n1)

#pragma unroll 2
    for (int kc = 0; kc < KC; ++kc) {
        const int k0 = kc * 32;
        // Prefetch next chunk one iteration ahead (speculative, L2-resident)
        if (kc + 1 < KC) {
            __builtin_prefetch(Acur + (size_t)arow0 * KP + k0 + 32, 0, 1);
            __builtin_prefetch(Wb + (size_t)(k0 + 32 + lane) * Hsz + nb, 0, 1);
        }
        // A fragments (16x32 bf16 each): lane holds K=[k0+asub,+8),[k0+16+asub,+8)
        const bf16_t* ap0 = Acur + (size_t)arow0 * KP + k0 + asub;
        v16bf af0 = load_frag(ap0, ap0 + 16);
        const bf16_t* ap1 = ap0 + (size_t)16 * KP;   // m-tile 1 = rows +16
        v16bf af1 = load_frag(ap1, ap1 + 16);
        // B fragments (32x16 bf16 each): lane holds col nb+(lane&15), K=[k0+asub,+8),[k0+16+asub,+8)
        const bf16_t* bp0 = Wb + (size_t)(nb + (lane & 15)) * KP + k0 + asub;
        v16bf bf0 = load_frag(bp0, bp0 + 16);
        v16bf bf1 = load_frag(bp0 + (size_t)16 * KP, bp0 + (size_t)16 * KP + 16);

        acc00 = __builtin_amdgcn_wmma_f32_16x16x32_bf16(
            false, af0, false, bf0, (short)0, acc00, false, false);
        acc01 = __builtin_amdgcn_wmma_f32_16x16x32_bf16(
            false, af0, false, bf1, (short)0, acc01, false, false);
        acc10 = __builtin_amdgcn_wmma_f32_16x16x32_bf16(
            false, af1, false, bf0, (short)0, acc10, false, false);
        acc11 = __builtin_amdgcn_wmma_f32_16x16x32_bf16(
            false, af1, false, bf1, (short)0, acc11, false, false);
    }

    // ---- bias add + bf16 store into Anext h-region (cols [E, E+H)) ----
    // C/D layout: VGPR r -> row tile_base + r + 8*(lane>>4), col nb + (lane&15)
    const int row0  = mbase + ((lane >> 4) ? 8 : 0);
    const int nc0   = nb + (lane & 15);
    const float bias0 = bias[nc0];
    const float bias1 = bias[nc0 + 16];
#pragma unroll
    for (int r = 0; r < 8; ++r) {
        Anext[(size_t)(row0 + r)      * KP + (Esz + nc0)]      = f2bf(acc00[r] + bias0);
        Anext[(size_t)(row0 + r)      * KP + (Esz + nc0 + 16)] = f2bf(acc01[r] + bias1);
        Anext[(size_t)(row0 + 16 + r) * KP + (Esz + nc0)]      = f2bf(acc10[r] + bias0);
        Anext[(size_t)(row0 + 16 + r) * KP + (Esz + nc0 + 16)] = f2bf(acc11[r] + bias1);
    }
}

// ---------------------------------------------------------------------------
// Output head: logits = Afin[:, 0:812] @ W_i2o^T + b_i2o; log_softmax over 3.
// One wave per batch row; wave-wide dot products with xor-shuffle reduction.
// ---------------------------------------------------------------------------
__global__ __launch_bounds__(256)
void rnn_out_kernel(const bf16_t* __restrict__ Afin,
                    const float* __restrict__ W_i2o,
                    const float* __restrict__ b_i2o,
                    float* __restrict__ out) {
    const int g    = (blockIdx.x * blockDim.x + threadIdx.x) >> 5;  // wave id = row
    const int lane = threadIdx.x & 31;
    if (g >= Bsz) return;

    float l[Osz];
#pragma unroll
    for (int o = 0; o < Osz; ++o) {
        float s = 0.0f;
        for (int k = lane; k < Ksz; k += 32)
            s += bf2f(Afin[(size_t)g * KP + k]) * W_i2o[(size_t)o * Ksz + k];
#pragma unroll
        for (int off = 16; off > 0; off >>= 1)
            s += __shfl_xor(s, off, 32);
        l[o] = s + b_i2o[o];
    }
    if (lane == 0) {
        float m = fmaxf(l[0], fmaxf(l[1], l[2]));
        float sum = expf(l[0] - m) + expf(l[1] - m) + expf(l[2] - m);
        float logz = m + logf(sum);
        out[g * 3 + 0] = l[0] - logz;
        out[g * 3 + 1] = l[1] - logz;
        out[g * 3 + 2] = l[2] - logz;
    }
}

// ---------------------------------------------------------------------------
// Launch: 1 init + 511 step kernels (ping-pong A buffers) + 1 head kernel.
// Workspace: Wb (512x832 bf16), A0, A1 (512x832 bf16 each) ~ 2.6 MB.
// ---------------------------------------------------------------------------
extern "C" void kernel_launch(void* const* d_in, const int* in_sizes, int n_in,
                              void* d_out, int out_size, void* d_ws, size_t ws_size,
                              hipStream_t stream) {
    (void)in_sizes; (void)n_in; (void)out_size; (void)ws_size;
    const int*   batch_x  = (const int*)  d_in[0];
    const float* emb_tab  = (const float*)d_in[1];
    const float* W_i2h    = (const float*)d_in[2];
    const float* b_i2h    = (const float*)d_in[3];
    const float* W_i2o    = (const float*)d_in[4];
    const float* b_i2o    = (const float*)d_in[5];
    float*       out      = (float*)d_out;

    char* ws = (char*)d_ws;
    const size_t WB_BYTES = (size_t)KP * Hsz * sizeof(bf16_t);
    const size_t A_BYTES  = (size_t)Bsz * KP * sizeof(bf16_t);
    auto align256 = [](size_t x) { return (x + 255) & ~(size_t)255; };
    bf16_t* Wb = (bf16_t*)ws;
    bf16_t* A0 = (bf16_t*)(ws + align256(WB_BYTES));
    bf16_t* A1 = (bf16_t*)(ws + align256(WB_BYTES) + align256(A_BYTES));

    rnn_init_kernel<<<512, 256, 0, stream>>>(batch_x, emb_tab, W_i2h, Wb, A0, A1);

    bf16_t* bufs[2] = {A0, A1};
    dim3 grid(4, 8), blk(256);   // N blocks x M blocks; 64x128 tile per block
    for (int t = 0; t < Tsz - 1; ++t) {
        rnn_step_kernel<<<grid, blk, 0, stream>>>(
            bufs[t & 1], bufs[(t + 1) & 1], Wb, b_i2h, batch_x, emb_tab, t + 1);
    }
    // (T-1) & 1 == 1 -> final combined state lives in A1
    rnn_out_kernel<<<64, 256, 0, stream>>>(bufs[(Tsz - 1) & 1], W_i2o, b_i2o, out);
}
